// FactorHRNNModel_79302276153839
// MI455X (gfx1250) — compile-verified, matrix operand order checked by reading, not run on hardware
//
#include <hip/hip_runtime.h>

// ---------------- problem constants ----------------
#define BSZ   64
#define TSZ   256
#define NINP  256
#define NHID  512
#define NG    2048            // 4*NHID gate width (permuted: p = 4*j + gate)
#define KLOW  1024            // recurrent K for both cells
#define NTOK0 2048
#define NTOK1 512
#define MROW  (TSZ*BSZ)       // 16384
#define RBLK  64              // persistent blocks for recurrence
#define HSTATE (BSZ*1024)     // one state operand buffer (u16 elems)

typedef unsigned short u16;
typedef __attribute__((ext_vector_type(16))) __bf16 v16bf;
typedef __attribute__((ext_vector_type(8)))  __bf16 v8bf;
typedef __attribute__((ext_vector_type(8)))  float  v8f;

// ---------------- workspace layout (bytes) ----------------
// Gate dimension of WX/WLOWR/WHIGHR/BLOW/BHIGH/GX is PERMUTED: p = 4*j + g,
// original column n = g*512 + j  (g: 0=i,1=f,2=g,3=o).
constexpr size_t OFF_WX     = 0;                                   // u16[2048*512]
constexpr size_t OFF_WLOWR  = OFF_WX     + (size_t)NG*512*2;       // u16[2048*1024] [Wih_low[:,512:] | Whh_low]
constexpr size_t OFF_WHIGHR = OFF_WLOWR  + (size_t)NG*1024*2;      // u16[2048*1024] [Wih_high | Whh_high]
constexpr size_t OFF_WDEC0  = OFF_WHIGHR + (size_t)NG*1024*2;      // u16[2048*512]
constexpr size_t OFF_WDEC1  = OFF_WDEC0  + (size_t)NTOK0*512*2;    // u16[512*512]
constexpr size_t OFF_BLOW   = OFF_WDEC1  + (size_t)NTOK1*512*2;    // f32[2048] permuted bih+bhh (low)
constexpr size_t OFF_BHIGH  = OFF_BLOW   + (size_t)NG*4;           // f32[2048] permuted (high)
constexpr size_t OFF_XS     = OFF_BHIGH  + (size_t)NG*4;           // u16[16384*512]
constexpr size_t OFF_GX     = OFF_XS     + (size_t)MROW*512*2;     // u16[16384*2048] permuted x-gates
constexpr size_t OFF_HS     = OFF_GX     + (size_t)MROW*NG*2;      // u16[16384*512] h_low per step
constexpr size_t OFF_HCAT   = OFF_HS     + (size_t)MROW*512*2;     // u16[2][64*1024] [Hh|Hl], parity buffered
constexpr size_t OFF_HCAT2  = OFF_HCAT   + (size_t)2*HSTATE*2;     // u16[2][64*1024] [Hl|Hh], parity buffered
constexpr size_t OFF_CL     = OFF_HCAT2  + (size_t)2*HSTATE*2;     // f32[64*512]
constexpr size_t OFF_CH     = OFF_CL     + (size_t)BSZ*NHID*4;     // f32[64*512]
constexpr size_t OFF_MASK   = OFF_CH     + (size_t)BSZ*NHID*4;     // f32[16384]
constexpr size_t OFF_BAR    = OFF_MASK   + (size_t)MROW*4;         // u32[2]

// ---------------- helpers ----------------
__device__ __forceinline__ u16 f2bf(float f) {
  unsigned u = __float_as_uint(f);
  unsigned r = 0x7FFFu + ((u >> 16) & 1u);
  return (u16)((u + r) >> 16);
}
__device__ __forceinline__ float bf2f(u16 h) { return __uint_as_float(((unsigned)h) << 16); }
__device__ __forceinline__ float sigm(float x) { return 1.f / (1.f + __expf(-x)); }

// One 16x16 tile, K-loop of v_wmma_f32_16x16x32_bf16.
// A: row-major M x K bf16.  W: row-major N x K bf16 (B = W^T).
// Per-lane K index = (e&7) + (e>>3)*16 + (lane>>4)*8 -> two contiguous 16B chunks.
__device__ __forceinline__ v8f wmma_tile(const u16* A, const u16* W, int K,
                                         int tm, int tn, int lane, v8f acc) {
  const __bf16* pa = (const __bf16*)A + (size_t)(tm * 16 + (lane & 15)) * K + ((lane >> 4) << 3);
  const __bf16* pb = (const __bf16*)W + (size_t)(tn * 16 + (lane & 15)) * K + ((lane >> 4) << 3);
  for (int k = 0; k < K; k += 32) {
    union { v16bf v; v8bf h[2]; } ua, ub;
    ua.h[0] = *(const v8bf*)(pa + k);
    ua.h[1] = *(const v8bf*)(pa + k + 16);
    ub.h[0] = *(const v8bf*)(pb + k);
    ub.h[1] = *(const v8bf*)(pb + k + 16);
    acc = __builtin_amdgcn_wmma_f32_16x16x32_bf16(false, ua.v, false, ub.v,
                                                  (short)0, acc, false, false);
  }
  return acc;
}

// software grid barrier (RBLK co-resident blocks, persistent kernel)
__device__ __forceinline__ void grid_bar(unsigned* bar) {
  __threadfence();
  __syncthreads();
  if (threadIdx.x == 0) {
    volatile unsigned* gen = bar + 1;
    unsigned g = *gen;
    unsigned old = atomicAdd(bar, 1u);
    if (old == (unsigned)(RBLK - 1)) {
      atomicExch(bar, 0u);
      __threadfence();
      atomicAdd((unsigned*)(bar + 1), 1u);
    } else {
      while (*gen == g) __builtin_amdgcn_s_sleep(2);
    }
  }
  __syncthreads();
}

// ---------------- kernels ----------------
__global__ __launch_bounds__(256) void prep_kernel(
    char* ws,
    const float* Wih_low, const float* bih_low, const float* Whh_low, const float* bhh_low,
    const float* Wih_high, const float* bih_high, const float* Whh_high, const float* bhh_high,
    const float* Wdec0, const float* Wdec1,
    const float* h_low0, const float* c_low0, const float* h_high0, const float* c_high0) {
  u16* WX     = (u16*)(ws + OFF_WX);
  u16* WLOWR  = (u16*)(ws + OFF_WLOWR);
  u16* WHIGHR = (u16*)(ws + OFF_WHIGHR);
  u16* WD0    = (u16*)(ws + OFF_WDEC0);
  u16* WD1    = (u16*)(ws + OFF_WDEC1);
  float* BLOW = (float*)(ws + OFF_BLOW);
  float* BHIGH= (float*)(ws + OFF_BHIGH);
  u16* HCAT   = (u16*)(ws + OFF_HCAT);
  u16* HCAT2  = (u16*)(ws + OFF_HCAT2);
  float* CL   = (float*)(ws + OFF_CL);
  float* CH   = (float*)(ws + OFF_CH);
  unsigned* BAR = (unsigned*)(ws + OFF_BAR);

  const int tid = blockIdx.x * blockDim.x + threadIdx.x;
  const int nt = gridDim.x * blockDim.x;

  for (int i = tid; i < NG * 512; i += nt) {
    int p = i >> 9, k = i & 511;
    int n = (p & 3) * 512 + (p >> 2);                 // gate-interleave permute
    WX[i] = f2bf(Wih_low[(size_t)n * 1024 + k]);
  }
  for (int i = tid; i < NG * 1024; i += nt) {
    int p = i >> 10, k = i & 1023;
    int n = (p & 3) * 512 + (p >> 2);
    float v = (k < 512) ? Wih_low[(size_t)n * 1024 + 512 + k]
                        : Whh_low[(size_t)n * 512 + (k - 512)];
    WLOWR[i] = f2bf(v);
  }
  for (int i = tid; i < NG * 1024; i += nt) {
    int p = i >> 10, k = i & 1023;
    int n = (p & 3) * 512 + (p >> 2);
    float v = (k < 512) ? Wih_high[(size_t)n * 512 + k]
                        : Whh_high[(size_t)n * 512 + (k - 512)];
    WHIGHR[i] = f2bf(v);
  }
  for (int i = tid; i < NTOK0 * 512; i += nt) WD0[i] = f2bf(Wdec0[i]);
  for (int i = tid; i < NTOK1 * 512; i += nt) WD1[i] = f2bf(Wdec1[i]);
  for (int i = tid; i < NG; i += nt) {
    int n = (i & 3) * 512 + (i >> 2);
    BLOW[i]  = bih_low[n] + bhh_low[n];
    BHIGH[i] = bih_high[n] + bhh_high[n];
  }
  for (int i = tid; i < BSZ * NHID; i += nt) {
    int b = i >> 9, j = i & 511;
    u16 hh = f2bf(h_high0[i]), hl = f2bf(h_low0[i]);
    HCAT [b * 1024 + j]                  = hh;
    HCAT [b * 1024 + 512 + j]            = hl;
    HCAT [HSTATE + b * 1024 + j]         = hh;
    HCAT [HSTATE + b * 1024 + 512 + j]   = hl;
    HCAT2[b * 1024 + j]                  = hl;
    HCAT2[b * 1024 + 512 + j]            = hh;
    HCAT2[HSTATE + b * 1024 + j]         = hl;
    HCAT2[HSTATE + b * 1024 + 512 + j]   = hh;
    CL[i] = c_low0[i];
    CH[i] = c_high0[i];
  }
  if (tid < 2) BAR[tid] = 0u;
}

__global__ __launch_bounds__(256) void embed_kernel(
    char* ws, const int* input0, const int* input1,
    const float* emb0, const float* emb1, const int* measure) {
  u16* XS = (u16*)(ws + OFF_XS);
  float* MASK = (float*)(ws + OFF_MASK);
  const int tid = blockIdx.x * blockDim.x + threadIdx.x;
  const int nt = gridDim.x * blockDim.x;
  for (int i = tid; i < MROW * 512; i += nt) {
    int row = i >> 9, k = i & 511;
    int t = row >> 6, b = row & 63;
    float v;
    if (k < NINP) v = emb0[(size_t)input0[b * TSZ + t] * NINP + k];
    else          v = emb1[(size_t)input1[b * TSZ + t] * NINP + (k - NINP)];
    XS[i] = f2bf(v);
  }
  const int mv = *measure;
  for (int i = tid; i < MROW; i += nt) {
    int t = i >> 6, b = i & 63;
    MASK[i] = (input0[b * TSZ + t] == mv) ? 1.0f : 0.0f;
  }
}

template <bool OUT_BF16, bool REMAP>
__device__ __forceinline__ void store_tile(void* outv, int N, int mbase, int nbase,
                                           int lane, v8f acc) {
  const int n = nbase + (lane & 15);
  const int m0 = mbase + ((lane >> 4) << 3);
  if (OUT_BF16) {
    u16* out = (u16*)outv;
    for (int r = 0; r < 8; ++r) out[(size_t)(m0 + r) * N + n] = f2bf(acc[r]);
  } else {
    float* out = (float*)outv;
    for (int r = 0; r < 8; ++r) {
      int m = m0 + r;
      size_t row = REMAP ? (size_t)((m & 63) * TSZ + (m >> 6)) : (size_t)m;
      out[row * N + n] = acc[r];
    }
  }
}

// 2x2 register-blocked WMMA GEMM: out[M,N] = A[M,K] * W[N,K]^T + bias[N]
// Each wave owns a 32x32 supertile: 8 b128 loads feed 4 WMMAs (2 loads/WMMA).
template <bool OUT_BF16, bool REMAP>
__global__ __launch_bounds__(256) void wmma_gemm2_kernel(
    const u16* __restrict__ A, const u16* __restrict__ W,
    const float* __restrict__ bias, void* outv, int M, int N, int K) {
  const int lane = threadIdx.x & 31;
  const int wpb = blockDim.x >> 5;
  int wt = blockIdx.x * wpb + (threadIdx.x >> 5);
  const int wstride = gridDim.x * wpb;
  const int mt = M >> 5, ntl = N >> 5;
  const int total = mt * ntl;
  for (; wt < total; wt += wstride) {
    const int tm = wt / ntl, tn = wt % ntl;
    const int koff = (lane >> 4) << 3;
    const __bf16* pa0 = (const __bf16*)A + (size_t)(tm * 32 + (lane & 15)) * K + koff;
    const __bf16* pa1 = pa0 + (size_t)16 * K;
    const __bf16* pb0 = (const __bf16*)W + (size_t)(tn * 32 + (lane & 15)) * K + koff;
    const __bf16* pb1 = pb0 + (size_t)16 * K;
    v8f acc00, acc01, acc10, acc11;
    {
      float b0 = bias[tn * 32 + (lane & 15)];
      float b1 = bias[tn * 32 + 16 + (lane & 15)];
      for (int r = 0; r < 8; ++r) { acc00[r] = b0; acc01[r] = b1; acc10[r] = b0; acc11[r] = b1; }
    }
    for (int k = 0; k < K; k += 32) {
      union { v16bf v; v8bf h[2]; } a0, a1, b0, b1;
      a0.h[0] = *(const v8bf*)(pa0 + k); a0.h[1] = *(const v8bf*)(pa0 + k + 16);
      a1.h[0] = *(const v8bf*)(pa1 + k); a1.h[1] = *(const v8bf*)(pa1 + k + 16);
      b0.h[0] = *(const v8bf*)(pb0 + k); b0.h[1] = *(const v8bf*)(pb0 + k + 16);
      b1.h[0] = *(const v8bf*)(pb1 + k); b1.h[1] = *(const v8bf*)(pb1 + k + 16);
      acc00 = __builtin_amdgcn_wmma_f32_16x16x32_bf16(false, a0.v, false, b0.v, (short)0, acc00, false, false);
      acc01 = __builtin_amdgcn_wmma_f32_16x16x32_bf16(false, a0.v, false, b1.v, (short)0, acc01, false, false);
      acc10 = __builtin_amdgcn_wmma_f32_16x16x32_bf16(false, a1.v, false, b0.v, (short)0, acc10, false, false);
      acc11 = __builtin_amdgcn_wmma_f32_16x16x32_bf16(false, a1.v, false, b1.v, (short)0, acc11, false, false);
    }
    store_tile<OUT_BF16, REMAP>(outv, N, tm * 32,      tn * 32,      lane, acc00);
    store_tile<OUT_BF16, REMAP>(outv, N, tm * 32,      tn * 32 + 16, lane, acc01);
    store_tile<OUT_BF16, REMAP>(outv, N, tm * 32 + 16, tn * 32,      lane, acc10);
    store_tile<OUT_BF16, REMAP>(outv, N, tm * 32 + 16, tn * 32 + 16, lane, acc11);
  }
}

// Persistent fused recurrence: 64 blocks x 8 wave32 = 512 waves = one per 16x16
// gate tile. Gate-interleaved weights let each wave finish the LSTM pointwise
// math via 4 cross-lane shuffles -> only 2 grid barriers/step, no gate traffic.
// State operands are parity double-buffered so no wave writes a buffer another
// wave is still streaming through WMMA.
__global__ __launch_bounds__(256, 1) void recurrent_kernel(char* ws, float* out_states) {
  const u16* GX     = (const u16*)(ws + OFF_GX);
  const u16* WLOWR  = (const u16*)(ws + OFF_WLOWR);
  const u16* WHIGHR = (const u16*)(ws + OFF_WHIGHR);
  u16* HS = (u16*)(ws + OFF_HS);
  u16* HCAT  = (u16*)(ws + OFF_HCAT);
  u16* HCAT2 = (u16*)(ws + OFF_HCAT2);
  float* CL = (float*)(ws + OFF_CL);
  float* CH = (float*)(ws + OFF_CH);
  const float* MASK  = (const float*)(ws + OFF_MASK);
  const float* BHIGH = (const float*)(ws + OFF_BHIGH);
  unsigned* BAR = (unsigned*)(ws + OFF_BAR);

  const int tid = blockIdx.x * blockDim.x + threadIdx.x;   // 0..16383
  const int lane = threadIdx.x & 31;
  const int wid = tid >> 5;                                // 0..511
  const int tn = wid >> 2;                                 // 0..127 (gate-tile col)
  const int tm = wid & 3;                                  // 0..3   (batch-tile row)
  const int ncol = tn * 16 + (lane & 15);                  // permuted gate column
  const int m0 = tm * 16 + ((lane >> 4) << 3);
  const int j = tn * 4 + ((lane & 15) >> 2);               // hidden unit
  const int sbase = lane & ~3;                             // lanes sbase..+3 = i,f,g,o
  const bool wr = (lane & 3) == 0;

  for (int t = 0; t < TSZ; ++t) {
    const int p = t & 1, q = p ^ 1;
    u16* Hc  = HCAT  + p * HSTATE;   // [Hh|Hl] current (read by low GEMM)
    u16* Hn  = HCAT  + q * HSTATE;   // next
    u16* H2c = HCAT2 + p * HSTATE;   // [Hl|Hh] current (read by high GEMM)
    u16* H2n = HCAT2 + q * HSTATE;   // next

    // ---- low cell: GEMM + fused gates ----
    {
      v8f acc;
      const u16* gx = GX + (size_t)t * BSZ * NG;
      for (int r = 0; r < 8; ++r) acc[r] = bf2f(gx[(size_t)(m0 + r) * NG + ncol]);
      acc = wmma_tile(Hc, WLOWR, KLOW, tm, tn, lane, acc);
      for (int r = 0; r < 8; ++r) {
        float vi = __shfl(acc[r], sbase + 0, 32);
        float vf = __shfl(acc[r], sbase + 1, 32);
        float vg = __shfl(acc[r], sbase + 2, 32);
        float vo = __shfl(acc[r], sbase + 3, 32);
        if (wr) {
          int m = m0 + r;
          int e = m * NHID + j;
          float c = sigm(vf) * CL[e] + sigm(vi) * tanhf(vg);
          float h = sigm(vo) * tanhf(c);
          CL[e] = c;
          u16 hb = f2bf(h);
          Hn [m * 1024 + 512 + j] = hb;                 // Hl for next step's low GEMM
          H2c[m * 1024 + j]       = hb;                 // Hl for this step's high GEMM
          HS[((size_t)t * BSZ + m) * NHID + j] = hb;    // decoder input
          if (t == TSZ - 1) { out_states[e] = h; out_states[BSZ * NHID + e] = c; }
        }
      }
    }
    grid_bar(BAR);

    // ---- high cell: GEMM + fused gates + masked state update ----
    {
      v8f acc;
      float bv = BHIGH[ncol];
      for (int r = 0; r < 8; ++r) acc[r] = bv;
      acc = wmma_tile(H2c, WHIGHR, KLOW, tm, tn, lane, acc);
      for (int r = 0; r < 8; ++r) {
        float vi = __shfl(acc[r], sbase + 0, 32);
        float vf = __shfl(acc[r], sbase + 1, 32);
        float vg = __shfl(acc[r], sbase + 2, 32);
        float vo = __shfl(acc[r], sbase + 3, 32);
        if (wr) {
          int m = m0 + r;
          int e = m * NHID + j;
          float chold = CH[e];
          float ct = sigm(vf) * chold + sigm(vi) * tanhf(vg);
          float ht = sigm(vo) * tanhf(ct);
          float mk = MASK[t * BSZ + m];
          float hold = bf2f(Hc[m * 1024 + j]);          // old Hh (stable this phase)
          float hnew = mk * ht + (1.f - mk) * hold;
          float cnew = mk * ct + (1.f - mk) * chold;
          CH[e] = cnew;
          u16 hb = f2bf(hnew);
          Hn [m * 1024 + j]       = hb;                 // Hh for next low GEMM
          H2n[m * 1024 + 512 + j] = hb;                 // Hh for next high GEMM
          if (t == TSZ - 1) {
            out_states[2 * BSZ * NHID + e] = hnew;
            out_states[3 * BSZ * NHID + e] = cnew;
          }
        }
      }
    }
    grid_bar(BAR);
  }
}

// ---------------- launch ----------------
extern "C" void kernel_launch(void* const* d_in, const int* in_sizes, int n_in,
                              void* d_out, int out_size, void* d_ws, size_t ws_size,
                              hipStream_t stream) {
  (void)in_sizes; (void)n_in; (void)out_size; (void)ws_size;
  const int*   input0   = (const int*)d_in[0];
  const int*   input1   = (const int*)d_in[1];
  const float* h_low0   = (const float*)d_in[2];
  const float* c_low0   = (const float*)d_in[3];
  const float* h_high0  = (const float*)d_in[4];
  const float* c_high0  = (const float*)d_in[5];
  const int*   measure  = (const int*)d_in[6];
  const float* emb0     = (const float*)d_in[7];
  const float* emb1     = (const float*)d_in[8];
  const float* Wih_low  = (const float*)d_in[9];
  const float* bih_low  = (const float*)d_in[10];
  const float* Whh_low  = (const float*)d_in[11];
  const float* bhh_low  = (const float*)d_in[12];
  const float* Wih_high = (const float*)d_in[13];
  const float* bih_high = (const float*)d_in[14];
  const float* Whh_high = (const float*)d_in[15];
  const float* bhh_high = (const float*)d_in[16];
  const float* Wdec0    = (const float*)d_in[17];
  const float* bdec0    = (const float*)d_in[18];
  const float* Wdec1    = (const float*)d_in[19];
  const float* bdec1    = (const float*)d_in[20];

  char* ws = (char*)d_ws;
  float* out = (float*)d_out;
  const size_t dec0_elems = (size_t)MROW * NTOK0;   // 33,554,432
  const size_t dec1_elems = (size_t)MROW * NTOK1;   //  8,388,608

  prep_kernel<<<512, 256, 0, stream>>>(ws, Wih_low, bih_low, Whh_low, bhh_low,
                                       Wih_high, bih_high, Whh_high, bhh_high,
                                       Wdec0, Wdec1, h_low0, c_low0, h_high0, c_high0);
  embed_kernel<<<1024, 256, 0, stream>>>(ws, input0, input1, emb0, emb1, measure);

  // Gx = xs @ Wih_low[:, :512]^T + (bih+bhh), gate-permuted N  [16384 x 2048] bf16
  wmma_gemm2_kernel<true, false><<<2048, 256, 0, stream>>>(
      (const u16*)(ws + OFF_XS), (const u16*)(ws + OFF_WX),
      (const float*)(ws + OFF_BLOW), (void*)(ws + OFF_GX), MROW, NG, 512);

  recurrent_kernel<<<RBLK, 256, 0, stream>>>(ws, out + dec0_elems + dec1_elems);

  // dec0 = h @ Wdec0^T + bdec0 -> d_out, (t,b)->(b,t) remap
  wmma_gemm2_kernel<false, true><<<2048, 256, 0, stream>>>(
      (const u16*)(ws + OFF_HS), (const u16*)(ws + OFF_WDEC0),
      bdec0, (void*)out, MROW, NTOK0, 512);
  // dec1 = h @ Wdec1^T + bdec1
  wmma_gemm2_kernel<false, true><<<1024, 256, 0, stream>>>(
      (const u16*)(ws + OFF_HS), (const u16*)(ws + OFF_WDEC1),
      bdec1, (void*)(out + dec0_elems), MROW, NTOK1, 512);
}
